// FinalAttentionModel_80315888435988
// MI455X (gfx1250) — compile-verified, matrix operand order checked by reading, not run on hardware
//
#include <hip/hip_runtime.h>
#include <math.h>

typedef __attribute__((ext_vector_type(16))) _Float16 v16h;
typedef __attribute__((ext_vector_type(8)))  _Float16 v8h;
typedef __attribute__((ext_vector_type(8)))  float    v8f;

// ---------------------------------------------------------------------------
// Fragment loaders. Per-lane 16-half fragment: halves 0..7 = K(klo..klo+7),
// halves 8..15 = K(klo+16..klo+23); klo = lhi*8 folded into the pointer.
// ---------------------------------------------------------------------------
__device__ __forceinline__ v16h frag_row_f32(const float* __restrict__ p)
{
    const float4* q = (const float4*)p;       // 16B-aligned for all our ldas
    float4 x0 = q[0], x1 = q[1], x2 = q[4], x3 = q[5];
    v16h r;
    r[0]  = (_Float16)x0.x; r[1]  = (_Float16)x0.y; r[2]  = (_Float16)x0.z; r[3]  = (_Float16)x0.w;
    r[4]  = (_Float16)x1.x; r[5]  = (_Float16)x1.y; r[6]  = (_Float16)x1.z; r[7]  = (_Float16)x1.w;
    r[8]  = (_Float16)x2.x; r[9]  = (_Float16)x2.y; r[10] = (_Float16)x2.z; r[11] = (_Float16)x2.w;
    r[12] = (_Float16)x3.x; r[13] = (_Float16)x3.y; r[14] = (_Float16)x3.z; r[15] = (_Float16)x3.w;
    return r;
}

__device__ __forceinline__ v16h frag_row_f16(const _Float16* __restrict__ p)
{
    const v8h* q = (const v8h*)p;             // 16B-aligned
    v8h lo = q[0];                            // halves 0..7
    v8h hi = q[2];                            // halves at +16
    v16h r;
    #pragma unroll
    for (int j = 0; j < 8; ++j) { r[j] = lo[j]; r[8 + j] = hi[j]; }
    return r;
}

__device__ __forceinline__ v16h frag_col_f32(const float* __restrict__ colbase, int ldb, int kbase)
{
    v16h r;
    #pragma unroll
    for (int j = 0; j < 8; ++j) r[j]     = (_Float16)colbase[(size_t)(kbase + j) * (size_t)ldb];
    #pragma unroll
    for (int j = 0; j < 8; ++j) r[8 + j] = (_Float16)colbase[(size_t)(kbase + 16 + j) * (size_t)ldb];
    return r;
}

// ---------------------------------------------------------------------------
// Batched WMMA GEMM:  C[z] = act( A[z] @ op(B[z]) + bias (+ Cprev) )
//   A: [M,K] row-major f32.
//   BT=1: B is [N,K] row-major (Linear weight). BH=1 -> B stored as f16.
//   BT=0: B is [K,N] row-major f32 (attn @ V path).
// Block: 128 threads = 4 waves; block tile 64(M) x 128(N); wave tile 32x64
// (2 M-tiles x 4 N-tiles => 8 WMMAs per 32-K step).
// All M % 64 == 0, N % 128 == 0, K % 32 == 0 in this model.
// ---------------------------------------------------------------------------
template <int BT, int BH>
__global__ __launch_bounds__(128)
void gemm_wmma_f16_t(const float* __restrict__ A, int lda, long long sAb, long long sAh,
                     const void* __restrict__ Bv, int ldb, long long sBb, long long sBh,
                     const float* __restrict__ bias,
                     float* __restrict__ C, int ldc, long long sCb, long long sCh,
                     int Kk, int Hdim, int actRelu, int accum)
{
    const int z  = blockIdx.z;
    const int bb = z / Hdim;
    const int hh = z % Hdim;
    A += (size_t)bb * (size_t)sAb + (size_t)hh * (size_t)sAh;
    C += (size_t)bb * (size_t)sCb + (size_t)hh * (size_t)sCh;
    const float*    Bf = (const float*)Bv    + (size_t)bb * (size_t)sBb + (size_t)hh * (size_t)sBh;
    const _Float16* Bh = (const _Float16*)Bv + (size_t)bb * (size_t)sBb + (size_t)hh * (size_t)sBh;

    const int wave = threadIdx.x >> 5;   // 0..3
    const int lane = threadIdx.x & 31;
    const int lrow = lane & 15;          // M-row (A) / N-col (B,C) within 16-tile
    const int lhi  = lane >> 4;          // selects K-halves (and M-halves of C)

    const int m0 = blockIdx.y * 64  + (wave >> 1) * 32;
    const int n0 = blockIdx.x * 128 + (wave & 1) * 64;

    v8f acc[2][4];
    #pragma unroll
    for (int i = 0; i < 2; ++i)
        #pragma unroll
        for (int j = 0; j < 4; ++j) acc[i][j] = {};

    // Streaming row pointers (advance by 32 elements per K-step).
    const float* ap[2];
    ap[0] = A + (size_t)(m0 + lrow) * (size_t)lda + lhi * 8;
    ap[1] = ap[0] + (size_t)16 * (size_t)lda;

    const float*    bpf[4];
    const _Float16* bph[4];
    #pragma unroll
    for (int j = 0; j < 4; ++j) {
        if (BT && BH)      bph[j] = Bh + (size_t)(n0 + j * 16 + lrow) * (size_t)ldb + lhi * 8;
        else if (BT)       bpf[j] = Bf + (size_t)(n0 + j * 16 + lrow) * (size_t)ldb + lhi * 8;
        else               bpf[j] = Bf + (size_t)(n0 + j * 16 + lrow);  // column base
    }

    for (int k0 = 0; k0 < Kk; k0 += 32) {
        __builtin_prefetch(ap[0] + 64, 0, 3);
        __builtin_prefetch(ap[1] + 64, 0, 3);

        v16h af[2];
        #pragma unroll
        for (int i = 0; i < 2; ++i) af[i] = frag_row_f32(ap[i]);

        v16h bf[4];
        if (BT && BH) {
            __builtin_prefetch(bph[0] + 64, 0, 3);
            __builtin_prefetch(bph[2] + 64, 0, 3);
            #pragma unroll
            for (int j = 0; j < 4; ++j) bf[j] = frag_row_f16(bph[j]);
        } else if (BT) {
            __builtin_prefetch(bpf[0] + 64, 0, 3);
            __builtin_prefetch(bpf[2] + 64, 0, 3);
            #pragma unroll
            for (int j = 0; j < 4; ++j) bf[j] = frag_row_f32(bpf[j]);
        } else {
            const int kbase = k0 + lhi * 8;
            #pragma unroll
            for (int j = 0; j < 4; ++j) bf[j] = frag_col_f32(bpf[j], ldb, kbase);
        }

        #pragma unroll
        for (int i = 0; i < 2; ++i)
            #pragma unroll
            for (int j = 0; j < 4; ++j)
                acc[i][j] = __builtin_amdgcn_wmma_f32_16x16x32_f16(
                    false, af[i], false, bf[j], (short)0, acc[i][j], false, false);

        ap[0] += 32; ap[1] += 32;
        if (BT && BH)      { bph[0] += 32; bph[1] += 32; bph[2] += 32; bph[3] += 32; }
        else if (BT)       { bpf[0] += 32; bpf[1] += 32; bpf[2] += 32; bpf[3] += 32; }
    }

    // Epilogue. D layout: lanes 0-15: N=lane, M=r; lanes 16-31: N=lane-16, M=8+r.
    #pragma unroll
    for (int i = 0; i < 2; ++i) {
        #pragma unroll
        for (int j = 0; j < 4; ++j) {
            const int col = n0 + j * 16 + lrow;
            const float bv = bias ? bias[col] : 0.0f;
            #pragma unroll
            for (int r = 0; r < 8; ++r) {
                const int row = m0 + i * 16 + lhi * 8 + r;
                float v = acc[i][j][r] + bv;
                if (accum)   v += C[(size_t)row * (size_t)ldc + col];
                if (actRelu) v = fmaxf(v, 0.0f);
                C[(size_t)row * (size_t)ldc + col] = v;
            }
        }
    }
}

// ---------------------------------------------------------------------------
// f32 -> f16 weight staging.
// ---------------------------------------------------------------------------
__global__ __launch_bounds__(256)
void cvt_f32_f16(const float* __restrict__ s, _Float16* __restrict__ d, long long n)
{
    long long i = (long long)blockIdx.x * 256 + threadIdx.x;
    if (i < n) d[i] = (_Float16)s[i];
}

// ---------------------------------------------------------------------------
// Row softmax with pre-scale (one 256-thread block per row).
// ---------------------------------------------------------------------------
__global__ __launch_bounds__(256)
void softmax_rows(float* __restrict__ S, int width, float scale)
{
    __shared__ float red[256];
    float* p = S + (size_t)blockIdx.x * (size_t)width;
    const int t = threadIdx.x;

    float m = -3.4e38f;
    for (int i = t; i < width; i += 256) m = fmaxf(m, p[i] * scale);
    red[t] = m; __syncthreads();
    for (int s = 128; s > 0; s >>= 1) { if (t < s) red[t] = fmaxf(red[t], red[t + s]); __syncthreads(); }
    m = red[0]; __syncthreads();

    float sum = 0.0f;
    for (int i = t; i < width; i += 256) { float e = expf(p[i] * scale - m); p[i] = e; sum += e; }
    red[t] = sum; __syncthreads();
    for (int s = 128; s > 0; s >>= 1) { if (t < s) red[t] += red[t + s]; __syncthreads(); }
    const float inv = 1.0f / red[0];
    for (int i = t; i < width; i += 256) p[i] *= inv;
}

// ---------------------------------------------------------------------------
// out = LayerNorm(X + R) * gamma + beta   (R may be null; one block per row)
// ---------------------------------------------------------------------------
__global__ __launch_bounds__(256)
void layernorm_res(const float* __restrict__ X, const float* __restrict__ R,
                   const float* __restrict__ g, const float* __restrict__ b,
                   float* __restrict__ out, int width)
{
    __shared__ float red[256];
    const size_t base = (size_t)blockIdx.x * (size_t)width;
    const int t = threadIdx.x;

    float s = 0.0f;
    for (int i = t; i < width; i += 256) {
        float v = X[base + i] + (R ? R[base + i] : 0.0f);
        s += v;
    }
    red[t] = s; __syncthreads();
    for (int k = 128; k > 0; k >>= 1) { if (t < k) red[t] += red[t + k]; __syncthreads(); }
    const float mean = red[0] / (float)width; __syncthreads();

    float sq = 0.0f;
    for (int i = t; i < width; i += 256) {
        float v = X[base + i] + (R ? R[base + i] : 0.0f) - mean;
        sq += v * v;
    }
    red[t] = sq; __syncthreads();
    for (int k = 128; k > 0; k >>= 1) { if (t < k) red[t] += red[t + k]; __syncthreads(); }
    const float inv = rsqrtf(red[0] / (float)width + 1e-5f);

    for (int i = t; i < width; i += 256) {
        float v = X[base + i] + (R ? R[base + i] : 0.0f);
        out[base + i] = (v - mean) * inv * g[i] + b[i];
    }
}

// gated = sig(G)*CTX + (1-sig(G))*Qin
__global__ __launch_bounds__(256)
void gate_mix(const float* __restrict__ G, const float* __restrict__ CTX,
              const float* __restrict__ Qin, float* __restrict__ out, long long n)
{
    long long i = (long long)blockIdx.x * 256 + threadIdx.x;
    if (i >= n) return;
    const float s = 1.0f / (1.0f + expf(-G[i]));
    out[i] = s * CTX[i] + (1.0f - s) * Qin[i];
}

// out = sig(G)*H   (out may alias G)
__global__ __launch_bounds__(256)
void gate_mul(const float* __restrict__ G, const float* __restrict__ H,
              float* __restrict__ out, long long n)
{
    long long i = (long long)blockIdx.x * 256 + threadIdx.x;
    if (i >= n) return;
    const float s = 1.0f / (1.0f + expf(-G[i]));
    out[i] = s * H[i];
}

__global__ __launch_bounds__(256)
void add_inplace(float* __restrict__ O, const float* __restrict__ X, long long n)
{
    long long i = (long long)blockIdx.x * 256 + threadIdx.x;
    if (i >= n) return;
    O[i] += X[i];
}

// out[row, 0:w]=a  [w:2w]=b  [2w:3w]=c  [3w:4w]=d
__global__ __launch_bounds__(256)
void concat4(const float* __restrict__ a, const float* __restrict__ b,
             const float* __restrict__ c, const float* __restrict__ d,
             float* __restrict__ out, long long rows, int w)
{
    long long i = (long long)blockIdx.x * 256 + threadIdx.x;
    long long total = rows * (long long)(4 * w);
    if (i >= total) return;
    int col = (int)(i % (long long)(4 * w));
    long long row = i / (long long)(4 * w);
    int seg = col / w, cc = col % w;
    const float* src = (seg == 0) ? a : (seg == 1) ? b : (seg == 2) ? c : d;
    out[i] = src[row * (long long)w + cc];
}

// ---------------------------------------------------------------------------
// Host orchestration
// ---------------------------------------------------------------------------
static inline void launch_gemm(hipStream_t st,
                               const float* A, int lda, long long sAb, long long sAh,
                               const void* Bm, int ldb, long long sBb, long long sBh,
                               int bTrans, int bHalf, const float* bias,
                               float* C, int ldc, long long sCb, long long sCh,
                               int M, int Nn, int Kk, int Z, int Hdim,
                               int relu, int accum)
{
    dim3 grid(Nn / 128, M / 64, Z), block(128);
    if (bTrans && bHalf)
        gemm_wmma_f16_t<1, 1><<<grid, block, 0, st>>>(A, lda, sAb, sAh, Bm, ldb, sBb, sBh,
                                                      bias, C, ldc, sCb, sCh, Kk, Hdim, relu, accum);
    else if (bTrans)
        gemm_wmma_f16_t<1, 0><<<grid, block, 0, st>>>(A, lda, sAb, sAh, Bm, ldb, sBb, sBh,
                                                      bias, C, ldc, sCb, sCh, Kk, Hdim, relu, accum);
    else
        gemm_wmma_f16_t<0, 0><<<grid, block, 0, st>>>(A, lda, sAb, sAh, Bm, ldb, sBb, sBh,
                                                      bias, C, ldc, sCb, sCh, Kk, Hdim, relu, accum);
}

extern "C" void kernel_launch(void* const* d_in, const int* in_sizes, int n_in,
                              void* d_out, int out_size, void* d_ws, size_t ws_size,
                              hipStream_t stream)
{
    (void)in_sizes; (void)n_in; (void)out_size; (void)ws_size;

    const int Bb = 8, Nseq = 1024, D = 512, Hh = 4, hd = 128;
    const int E = 2048, hdF = 512, E3 = 3 * E, E4 = 4 * E;
    const long long BN = (long long)Bb * Nseq;          // 8192
    const long long LBN = BN;

    const float* x1in = (const float*)d_in[0];
    const float* x2in = (const float*)d_in[1];
    auto P = [&](int i) -> const float* { return (const float*)d_in[i]; };
    // cross layer l params at 2 + 14*l : wq bq wk bk wv bv wgq bgq wgc bgc wo bo ng nb
    const int FP = 2 + 14 * 3;  // = 44, final params start
    // 44:we 45:be 46:win 47:bin 48:wop 49:bop 50:n1g 51:n1b 52:wg 53:bg
    // 54:wf1 55:bf1 56:wf2 57:bf2 58:n2g 59:n2b 60:wout 61:bout

    float* ws = (float*)d_ws;
    size_t o = 0;
    float* S    = ws + o; o += (size_t)Bb * Hh * Nseq * Nseq;       // 33.55M
    float* X1A  = ws + o; o += (size_t)BN * D;
    float* X2A  = ws + o; o += (size_t)BN * D;
    float* X1B  = ws + o; o += (size_t)BN * D;
    float* X2B  = ws + o; o += (size_t)BN * D;
    float* Qb   = ws + o; o += (size_t)BN * D;
    float* Kb   = ws + o; o += (size_t)BN * D;
    float* Vb   = ws + o; o += (size_t)BN * D;
    float* CTX  = ws + o; o += (size_t)BN * D;
    float* Gb   = ws + o; o += (size_t)BN * D;
    float* GAT  = ws + o; o += (size_t)BN * D;
    float* FUS  = ws + o; o += (size_t)BN * D;
    float* F1   = ws;     // aliases S..FUS (all dead by FFN time); needs 67.1M < 79.7M
    float* XC   = ws + o; o += (size_t)BN * E;
    float* Hbuf = ws + o; o += (size_t)BN * E;
    float* QKV  = ws + o; o += (size_t)BN * E3;
    float* CTXF = ws + o; o += (size_t)BN * E;
    float* T1   = ws + o; o += (size_t)BN * E;
    float* H2   = ws + o; o += (size_t)BN * E;
    float* GF   = ws + o; o += (size_t)BN * E;
    float* F2   = ws + o; o += (size_t)BN * E;

    // ---- f16 weight staging region ----
    _Float16* wbase = (_Float16*)(ws + o);
    _Float16* wh[64] = {};
    size_t ho = 0;
    auto ew_grid = [](long long n) { return dim3((unsigned)((n + 255) / 256)); };
    auto cvtw = [&](int idx, long long n) {
        wh[idx] = wbase + ho;
        cvt_f32_f16<<<ew_grid(n), dim3(256), 0, stream>>>(P(idx), wh[idx], n);
        ho += (size_t)n;
    };
    for (int l = 0; l < 3; ++l) {
        const int pb = 2 + 14 * l;
        cvtw(pb + 0, (long long)D * D);   // wq
        cvtw(pb + 2, (long long)D * D);   // wk
        cvtw(pb + 4, (long long)D * D);   // wv
        cvtw(pb + 6, (long long)D * D);   // wgq
        cvtw(pb + 8, (long long)D * D);   // wgc
        cvtw(pb + 10, (long long)D * D);  // wo
    }
    cvtw(FP + 0,  (long long)E * E);      // we
    cvtw(FP + 2,  (long long)E3 * E);     // win
    cvtw(FP + 4,  (long long)E * E);      // wop
    cvtw(FP + 8,  (long long)E * E);      // wg
    cvtw(FP + 10, (long long)E4 * E);     // wf1
    cvtw(FP + 12, (long long)E * E4);     // wf2
    cvtw(FP + 16, (long long)E * E);      // wout

    const float scaleC = 1.0f / sqrtf((float)hd);
    const float scaleF = 1.0f / sqrtf((float)hdF);

    // ---- one cross-attention direction: q_in=qx, kv; writes `out` ----
    auto cross_dir = [&](const float* qx, const float* kv, int pb, float* out) {
        // Q,K,V projections (f16 weights)
        launch_gemm(stream, qx, D, 0, 0, wh[pb + 0], D, 0, 0, 1, 1, P(pb + 1),
                    Qb, D, 0, 0, (int)BN, D, D, 1, 1, 0, 0);
        launch_gemm(stream, kv, D, 0, 0, wh[pb + 2], D, 0, 0, 1, 1, P(pb + 3),
                    Kb, D, 0, 0, (int)BN, D, D, 1, 1, 0, 0);
        launch_gemm(stream, kv, D, 0, 0, wh[pb + 4], D, 0, 0, 1, 1, P(pb + 5),
                    Vb, D, 0, 0, (int)BN, D, D, 1, 1, 0, 0);
        // scores[b,h] = Q[b,h] @ K[b,h]^T  (f32 B)
        launch_gemm(stream,
                    Qb, D, (long long)Nseq * D, hd,
                    Kb, D, (long long)Nseq * D, hd, 1, 0, nullptr,
                    S, Nseq, (long long)Hh * Nseq * Nseq, (long long)Nseq * Nseq,
                    Nseq, Nseq, hd, Bb * Hh, Hh, 0, 0);
        softmax_rows<<<dim3(Bb * Hh * Nseq), dim3(256), 0, stream>>>(S, Nseq, scaleC);
        // ctx[b,h] = P @ V[b,h]   (B non-transposed f32)
        launch_gemm(stream,
                    S, Nseq, (long long)Hh * Nseq * Nseq, (long long)Nseq * Nseq,
                    Vb, D, (long long)Nseq * D, hd, 0, 0, nullptr,
                    CTX, D, (long long)Nseq * D, hd,
                    Nseq, hd, Nseq, Bb * Hh, Hh, 0, 0);
        // gate = wgq(qx)+bgq + wgc(ctx)+bgc
        launch_gemm(stream, qx, D, 0, 0, wh[pb + 6], D, 0, 0, 1, 1, P(pb + 7),
                    Gb, D, 0, 0, (int)BN, D, D, 1, 1, 0, 0);
        launch_gemm(stream, CTX, D, 0, 0, wh[pb + 8], D, 0, 0, 1, 1, P(pb + 9),
                    Gb, D, 0, 0, (int)BN, D, D, 1, 1, 0, 1);
        gate_mix<<<ew_grid(LBN * D), dim3(256), 0, stream>>>(Gb, CTX, qx, GAT, LBN * D);
        launch_gemm(stream, GAT, D, 0, 0, wh[pb + 10], D, 0, 0, 1, 1, P(pb + 11),
                    FUS, D, 0, 0, (int)BN, D, D, 1, 1, 0, 0);
        layernorm_res<<<dim3((unsigned)BN), dim3(256), 0, stream>>>(
            qx, FUS, P(pb + 12), P(pb + 13), out, D);
    };

    // ---- 3 cross layers, ping-pong ----
    const float* c1 = x1in; const float* c2 = x2in;
    float* o1 = X1A; float* o2 = X2A; float* a1 = X1B; float* a2 = X2B;
    for (int l = 0; l < 3; ++l) {
        const int pb = 2 + 14 * l;
        cross_dir(c1, c2, pb, o1);
        cross_dir(c2, c1, pb, o2);
        c1 = o1; c2 = o2;
        float* t;
        t = o1; o1 = a1; a1 = t;
        t = o2; o2 = a2; a2 = t;
    }

    // ---- final block ----
    concat4<<<ew_grid(LBN * E), dim3(256), 0, stream>>>(c1, x1in, c2, x2in, XC, LBN, D);

    // h = xcat @ we^T + be
    launch_gemm(stream, XC, E, 0, 0, wh[FP + 0], E, 0, 0, 1, 1, P(FP + 1),
                Hbuf, E, 0, 0, (int)BN, E, E, 1, 1, 0, 0);
    // qkv = h @ win^T + bin
    launch_gemm(stream, Hbuf, E, 0, 0, wh[FP + 2], E, 0, 0, 1, 1, P(FP + 3),
                QKV, E3, 0, 0, (int)BN, E3, E, 1, 1, 0, 0);
    // scores[b,h] = Q @ K^T  (slices of qkv, row stride 3E, f32 B)
    launch_gemm(stream,
                QKV, E3, (long long)Nseq * E3, hdF,
                QKV + E, E3, (long long)Nseq * E3, hdF, 1, 0, nullptr,
                S, Nseq, (long long)Hh * Nseq * Nseq, (long long)Nseq * Nseq,
                Nseq, Nseq, hdF, Bb * Hh, Hh, 0, 0);
    softmax_rows<<<dim3(Bb * Hh * Nseq), dim3(256), 0, stream>>>(S, Nseq, scaleF);
    // ctx[b,h] = P @ V
    launch_gemm(stream,
                S, Nseq, (long long)Hh * Nseq * Nseq, (long long)Nseq * Nseq,
                QKV + 2 * E, E3, (long long)Nseq * E3, hdF, 0, 0, nullptr,
                CTXF, E, (long long)Nseq * E, hdF,
                Nseq, hdF, Nseq, Bb * Hh, Hh, 0, 0);
    // attn_out = ctx @ wop^T + bop
    launch_gemm(stream, CTXF, E, 0, 0, wh[FP + 4], E, 0, 0, 1, 1, P(FP + 5),
                T1, E, 0, 0, (int)BN, E, E, 1, 1, 0, 0);
    // h2 = LN(h + attn_out)
    layernorm_res<<<dim3((unsigned)BN), dim3(256), 0, stream>>>(
        Hbuf, T1, P(FP + 6), P(FP + 7), H2, E);
    // gate = sigmoid(h2 @ wg^T + bg); gated = gate * h2
    launch_gemm(stream, H2, E, 0, 0, wh[FP + 8], E, 0, 0, 1, 1, P(FP + 9),
                GF, E, 0, 0, (int)BN, E, E, 1, 1, 0, 0);
    gate_mul<<<ew_grid(LBN * E), dim3(256), 0, stream>>>(GF, H2, GF, LBN * E);
    // ffn = wf2( relu( wf1(gated) ) )     (F1 aliases dead cross region)
    launch_gemm(stream, GF, E, 0, 0, wh[FP + 10], E, 0, 0, 1, 1, P(FP + 11),
                F1, E4, 0, 0, (int)BN, E4, E, 1, 1, 1, 0);
    launch_gemm(stream, F1, E4, 0, 0, wh[FP + 12], E4, 0, 0, 1, 1, P(FP + 13),
                F2, E, 0, 0, (int)BN, E, E4, 1, 1, 0, 0);
    // out = LN(gated + ffn)  -> reuse Hbuf
    layernorm_res<<<dim3((unsigned)BN), dim3(256), 0, stream>>>(
        GF, F2, P(FP + 14), P(FP + 15), Hbuf, E);
    // y = out @ wout^T + bout  -> d_out, then += shortcut (xcat)
    launch_gemm(stream, Hbuf, E, 0, 0, wh[FP + 16], E, 0, 0, 1, 1, P(FP + 17),
                (float*)d_out, E, 0, 0, (int)BN, E, E, 1, 1, 0, 0);
    add_inplace<<<ew_grid(LBN * E), dim3(256), 0, stream>>>((float*)d_out, XC, LBN * E);
}
